// FineRec_9552007266421
// MI455X (gfx1250) — compile-verified, matrix operand order checked by reading, not run on hardware
//
#include <hip/hip_runtime.h>

typedef _Float16 f16;
typedef __attribute__((ext_vector_type(16))) _Float16 v16h;
typedef __attribute__((ext_vector_type(8)))  _Float16 v8h;
typedef __attribute__((ext_vector_type(4)))  _Float16 v4h;
typedef __attribute__((ext_vector_type(8)))  float    v8f;

#define DIM   128   // chunk_embSize
#define ROWS  64    // padded sequence tile (L=50 -> 64)
#define XSTR  136   // LDS row stride in halves: 272B -> rows shift 4 banks, conflict-free b128 frag loads
#define VSTR  72    // transposed V stride in halves: 144B -> 36-bank stride, conflict-free b16 stores

// CDNA5 native v_tanh_f32 (confirmed in codegen); branchless fallback otherwise.
__device__ __forceinline__ float fast_tanh(float x) {
#if __has_builtin(__builtin_amdgcn_tanhf)
  return __builtin_amdgcn_tanhf(x);
#else
  float t = __builtin_amdgcn_exp2f(x * 2.88539008177793f);
  return 1.0f - 2.0f * __builtin_amdgcn_rcpf(t + 1.0f);
#endif
}

// Build a 16-bit WMMA A/B fragment (wave32 layout, ISA 7.12.2):
//   caller passes p = rowptr + baseK + h*8 (h = lane>>4)
//   elems 0..7  <- K = baseK + h*8 + 0..7        == p[0..7]
//   elems 8..15 <- K = baseK + 16 + h*8 + 0..7   == p[16..23]
__device__ __forceinline__ v16h make_frag(const f16* p) {
  v8h lo = *(const v8h*)(p);
  v8h hi = *(const v8h*)(p + 16);
  v16h r;
#pragma unroll
  for (int i = 0; i < 8; ++i) { r[i] = lo[i]; r[i + 8] = hi[i]; }
  return r;
}

// D = tanh(X[64x128] * W^T + bias); X f16 in LDS, W f16 row-major [DIM][DIM].
// TRANSPOSE=false: Out[row][col] (stride XSTR).  TRANSPOSE=true: Out[col][row] (stride VSTR).
template <bool TRANSPOSE>
__device__ __forceinline__ void gemm_tanh_store(
    const f16* __restrict__ X, const f16* __restrict__ W16,
    const float* __restrict__ bias, f16* __restrict__ Out,
    int wave, int lane)
{
  const int colbase = wave * 16;
  const int ln = lane & 15;
  const int h  = lane >> 4;
  v16h bf[4];
#pragma unroll
  for (int ks = 0; ks < 4; ++ks)   // B frags fixed per wave: column n == W row (colbase+ln)
    bf[ks] = make_frag(W16 + (size_t)(colbase + ln) * DIM + ks * 32 + h * 8);
  const float bb = bias[colbase + ln];
#pragma unroll
  for (int rt = 0; rt < 4; ++rt) {
    const int rowbase = rt * 16;
    v8f acc = {};
#pragma unroll
    for (int ks = 0; ks < 4; ++ks) {
      v16h a = make_frag(X + (rowbase + ln) * XSTR + ks * 32 + h * 8);
      acc = __builtin_amdgcn_wmma_f32_16x16x32_f16(false, a, false, bf[ks],
                                                   (short)0, acc, false, false);
    }
    // C/D layout: VGPR i -> M = i + 8*h, N = ln
#pragma unroll
    for (int i = 0; i < 8; ++i) {
      const int row = rowbase + h * 8 + i;
      const f16 v = (f16)fast_tanh(acc[i] + bb);
      if (TRANSPOSE) Out[(colbase + ln) * VSTR + row] = v;
      else           Out[row * XSTR + colbase + ln]   = v;
    }
  }
}

__global__ void finerec_cvt(const float* __restrict__ a, const float* __restrict__ b,
                            const float* __restrict__ c,
                            f16* __restrict__ oa, f16* __restrict__ ob,
                            f16* __restrict__ oc, int n)
{
  int i = blockIdx.x * blockDim.x + threadIdx.x;
  if (i < n) { oa[i] = (f16)a[i]; ob[i] = (f16)b[i]; oc[i] = (f16)c[i]; }
}

// q = tanh(user_emb @ Wq^T + bq), 64 users per 256-thread block, f32 out.
__global__ __launch_bounds__(256)
void finerec_q(const float* __restrict__ user_emb,
               const f16* __restrict__ Wq16, const float* __restrict__ bq,
               float* __restrict__ qbuf, int U)
{
  __shared__ f16 X[ROWS * XSTR];
  const int tid = threadIdx.x;
  const int ub  = blockIdx.x * ROWS;
  {
    const int r = tid >> 2, c0 = (tid & 3) * 32;
    const int u = ub + r;
    if (u < U) {
      const float* src = user_emb + (size_t)u * DIM;
#pragma unroll
      for (int c = c0; c < c0 + 32; c += 4) {
        v4h hx;
#pragma unroll
        for (int j = 0; j < 4; ++j) hx[j] = (f16)src[c + j];
        *(v4h*)&X[r * XSTR + c] = hx;
      }
    } else {
      const v4h z = {};
#pragma unroll
      for (int c = c0; c < c0 + 32; c += 4) *(v4h*)&X[r * XSTR + c] = z;
    }
  }
  __syncthreads();

  const int wave = tid >> 5, lane = tid & 31;
  const int colbase = wave * 16;
  const int ln = lane & 15, h = lane >> 4;
  v16h bf[4];
#pragma unroll
  for (int ks = 0; ks < 4; ++ks)
    bf[ks] = make_frag(Wq16 + (size_t)(colbase + ln) * DIM + ks * 32 + h * 8);
  const float bb = bq[colbase + ln];
  const bool full = (ub + ROWS) <= U;   // fast path: whole tile in range
#pragma unroll
  for (int rt = 0; rt < 4; ++rt) {
    v8f acc = {};
#pragma unroll
    for (int ks = 0; ks < 4; ++ks) {
      v16h a = make_frag(X + (rt * 16 + ln) * XSTR + ks * 32 + h * 8);
      acc = __builtin_amdgcn_wmma_f32_16x16x32_f16(false, a, false, bf[ks],
                                                   (short)0, acc, false, false);
    }
    float* dst = qbuf + (size_t)(ub + rt * 16 + h * 8) * DIM + colbase + ln;
    if (full) {
#pragma unroll
      for (int i = 0; i < 8; ++i) dst[i * DIM] = fast_tanh(acc[i] + bb);
    } else {
#pragma unroll
      for (int i = 0; i < 8; ++i) {
        if (ub + rt * 16 + h * 8 + i < U) dst[i * DIM] = fast_tanh(acc[i] + bb);
      }
    }
  }
}

// One block per user: gather -> two WMMA GEMMs (k,v) -> masked attention -> out.
__global__ __launch_bounds__(256)
void finerec_fused(const float* __restrict__ item_table,
                   const float* __restrict__ opin_table,
                   const float* __restrict__ attr,
                   const f16* __restrict__ Wk16, const float* __restrict__ bk,
                   const f16* __restrict__ Wv16, const float* __restrict__ bv,
                   const float* __restrict__ qbuf,
                   const int* __restrict__ iseq, const int* __restrict__ oseq,
                   float* __restrict__ out, int L)
{
  __shared__ f16 XK[ROWS * XSTR];
  __shared__ f16 XV[ROWS * XSTR];
  __shared__ f16 KM[ROWS * XSTR];      // k, row-major (rows read contiguously by q.k)
  __shared__ f16 VMT[DIM * VSTR];      // v, TRANSPOSED (dims read contiguously by w.v)
  __shared__ float qs[DIM];
  __shared__ float wpart[ROWS * 4];
  __shared__ float wbuf[ROWS];

  const int u   = blockIdx.x;
  const int tid = threadIdx.x;

  if (tid < DIM) qs[tid] = qbuf[(size_t)u * DIM + tid];

  { // gather item/opin rows, build f16 GEMM operands in LDS (4 threads/row x 32 cols)
    const int r  = tid >> 2;
    const int c0 = (tid & 3) * 32;
    if (r < L) {
      const int it = iseq[u * L + r];
      const int oi = oseq[u * L + r];
      const float* ip = item_table + (size_t)it * DIM;
      const float* pp = opin_table + (size_t)oi * DIM;
#pragma unroll
      for (int c = c0; c < c0 + 32; c += 4) {
        v4h hk, hv;
#pragma unroll
        for (int j = 0; j < 4; ++j) {
          const float iv = ip[c + j];
          hk[j] = (f16)(attr[c + j] + iv);
          hv[j] = (f16)(iv + pp[c + j]);
        }
        *(v4h*)&XK[r * XSTR + c] = hk;
        *(v4h*)&XV[r * XSTR + c] = hv;
      }
    } else { // padding rows -> zero (attention never reads them: wbuf==0 there)
      const v4h z = {};
#pragma unroll
      for (int c = c0; c < c0 + 32; c += 4) {
        *(v4h*)&XK[r * XSTR + c] = z;
        *(v4h*)&XV[r * XSTR + c] = z;
      }
    }
  }
  __syncthreads();

  const int wave = tid >> 5, lane = tid & 31;
  gemm_tanh_store<false>(XK, Wk16, bk, KM,  wave, lane); // k = tanh((attr+item)Wk^T+bk)
  gemm_tanh_store<true >(XV, Wv16, bv, VMT, wave, lane); // v = tanh((item+opin)Wv^T+bv), transposed
  __syncthreads();

  { // w[l] = mask * <q, k_l>: 4 threads/row, vectorized v8h row reads
    const int r = tid >> 2, seg = tid & 3;
    const f16*  kr = &KM[r * XSTR + seg * 32];
    const float* qp = &qs[seg * 32];
    float s = 0.f;
#pragma unroll
    for (int c = 0; c < 32; c += 8) {
      v8h kv = *(const v8h*)(kr + c);
#pragma unroll
      for (int j = 0; j < 8; ++j) s += qp[c + j] * (float)kv[j];
    }
    wpart[tid] = s;
  }
  __syncthreads();
  if (tid < ROWS) {
    float s = wpart[tid * 4] + wpart[tid * 4 + 1] + wpart[tid * 4 + 2] + wpart[tid * 4 + 3];
    if (!(tid < L && iseq[u * L + tid] > 0)) s = 0.f;  // padding mask (+ rows >= L)
    wbuf[tid] = s;
  }
  __syncthreads();

  if (tid < DIM) {   // out[d] = sum_l w[l] * v[l,d]; VMT row d is contiguous
    const f16* vr = &VMT[tid * VSTR];
    float s = 0.f;
#pragma unroll
    for (int c = 0; c < ROWS; c += 8) {   // rows >= L contribute 0 via wbuf
      v8h vv = *(const v8h*)(vr + c);
#pragma unroll
      for (int j = 0; j < 8; ++j) s += wbuf[c + j] * (float)vv[j];
    }
    out[(size_t)u * DIM + tid] = s;
  }
}

extern "C" void kernel_launch(void* const* d_in, const int* in_sizes, int n_in,
                              void* d_out, int out_size, void* d_ws, size_t ws_size,
                              hipStream_t stream)
{
  const float* item_table = (const float*)d_in[0];
  const float* opin_table = (const float*)d_in[1];
  const float* user_emb   = (const float*)d_in[2];
  const float* attr       = (const float*)d_in[3];
  const float* Wq         = (const float*)d_in[4];
  const float* bq         = (const float*)d_in[5];
  const float* Wk         = (const float*)d_in[6];
  const float* bk         = (const float*)d_in[7];
  const float* Wv         = (const float*)d_in[8];
  const float* bv         = (const float*)d_in[9];
  const int*   iseq       = (const int*)d_in[10];
  const int*   oseq       = (const int*)d_in[11];

  const int U = in_sizes[2] / DIM;
  int L = (U > 0) ? in_sizes[10] / U : 0;
  if (L > ROWS) L = ROWS;

  // Workspace layout: 3 f16 weight matrices (32 KB each) + q f32 [U,128]
  char* ws = (char*)d_ws;
  f16*   Wq16 = (f16*)(ws);
  f16*   Wk16 = (f16*)(ws + 1 * DIM * DIM * sizeof(f16));
  f16*   Wv16 = (f16*)(ws + 2 * DIM * DIM * sizeof(f16));
  float* qbuf = (float*)(ws + 3 * DIM * DIM * sizeof(f16));

  finerec_cvt<<<(DIM * DIM + 255) / 256, 256, 0, stream>>>(
      Wq, Wk, Wv, Wq16, Wk16, Wv16, DIM * DIM);

  finerec_q<<<(U + ROWS - 1) / ROWS, 256, 0, stream>>>(
      user_emb, Wq16, bq, qbuf, U);

  finerec_fused<<<U, 256, 0, stream>>>(
      item_table, opin_table, attr, Wk16, bk, Wv16, bv,
      qbuf, iseq, oseq, (float*)d_out, L);
}
